// TfmrAttention_38062000177348
// MI455X (gfx1250) — matrix operand order repaired and checked by reading, not compiled, not run on hardware
//
#include <hip/hip_runtime.h>
#include <hip/hip_bf16.h>

// ---------------------------------------------------------------------------
// GPT-2 attention block for MI455X (gfx1250, wave32, WMMA f16->f32).
//   K0: weight prep: c_attn_w / c_proj_w (f32 [K][N]) -> f16 [N][K] (K-major)
//   K1: qkv = hs @ c_attn_w + b  -> Q,K [B,H,S,Dh] f16, Vt [B,H,Dh,S] f16
//       (32x64 tile per wave, B-operand = two 8-half loads from f16 Wt)
//   K2: per (b,h,16-row block): scores=QK^T/8 (+causal) in LDS (128KB),
//       softmax, write attn_weights f32 once, causal-pruned PV -> f16 out
//   K3: out = attn_out @ c_proj_w + b (f32), same GEMM structure as K1
// ---------------------------------------------------------------------------

#define BATCH   2
#define SEQ     2048
#define DMODEL  1024
#define NHEADS  16
#define HEADDIM 64
#define MROWS   (BATCH * SEQ)       // 4096
#define BH      (BATCH * NHEADS)    // 32

typedef __attribute__((ext_vector_type(16))) _Float16 v16h;
typedef __attribute__((ext_vector_type(8)))  float    v8f;

#define WMMA_F16(a, b, c) \
  __builtin_amdgcn_wmma_f32_16x16x32_f16(false, (a), false, (b), (short)0, (c), false, false)

// ---- WMMA operand loaders (wave32 layouts per CDNA5 ISA 7.12.2) -----------
// A (16x32 f16): lanes 0-15 row M=lane, K = {k0..k0+7, k0+16..k0+23};
//                lanes 16-31 row M=lane-16, same +8.
__device__ __forceinline__ v16h load_a_f16(const _Float16* rowp, int k0, bool hi) {
  v16h a;
  const int o = k0 + (hi ? 8 : 0);
#pragma unroll
  for (int j = 0; j < 8; ++j) { a[j] = rowp[o + j]; a[j + 8] = rowp[o + 16 + j]; }
  return a;
}
__device__ __forceinline__ v16h load_a_f32cvt(const float* rowp, int k0, bool hi) {
  v16h a;
  const int o = k0 + (hi ? 8 : 0);
#pragma unroll
  for (int j = 0; j < 8; ++j) {
    a[j]     = (_Float16)rowp[o + j];
    a[j + 8] = (_Float16)rowp[o + 16 + j];
  }
  return a;
}
// B (32x16 f16): column N = lane&15; lanes<16 hold K={k0..k0+7, k0+16..k0+23},
// lanes>=16 same +8. Source stored K-contiguous per column (ld halfs per column).
__device__ __forceinline__ v16h load_b_kmajor(const _Float16* base, int ld, int col,
                                              int k0, bool hi) {
  typedef __attribute__((ext_vector_type(8))) _Float16 v8h_t;
  const _Float16* p = base + (size_t)col * ld + k0 + (hi ? 8 : 0);
  const v8h_t lo = *(const v8h_t*)p;
  const v8h_t up = *(const v8h_t*)(p + 16);
  return __builtin_shufflevector(lo, up, 0, 1, 2, 3, 4, 5, 6, 7, 8, 9, 10, 11, 12, 13, 14, 15);
}

// ---------------------------------------------------------------------------
// Kernel 0: weight prep.  W f32 [K][N] row-major -> Wt f16 [N][K] (K-major).
// 64x64 LDS tile transpose, coalesced on both sides.
// ---------------------------------------------------------------------------
__global__ __launch_bounds__(256) void wprep_kernel(const float* __restrict__ W,
                                                    _Float16* __restrict__ Wt,
                                                    int N, int K) {
  __shared__ _Float16 t[64][65];                 // pad to avoid bank conflicts
  const int tx = threadIdx.x;                    // 0..63
  const int ty = threadIdx.y;                    // 0..3
  const int n0 = blockIdx.x * 64;
  const int k0 = blockIdx.y * 64;
#pragma unroll
  for (int kk = ty; kk < 64; kk += 4)
    t[kk][tx] = (_Float16)W[(size_t)(k0 + kk) * N + n0 + tx];
  __syncthreads();
#pragma unroll
  for (int nn = ty; nn < 64; nn += 4)
    Wt[(size_t)(n0 + nn) * K + k0 + tx] = t[tx][nn];
}

// ---------------------------------------------------------------------------
// Kernel 1: QKV projection.  M=4096, N=3072, K=1024.  One wave -> 32x64 tile
// (2 M-subtiles reuse each B operand).
// ---------------------------------------------------------------------------
__global__ __launch_bounds__(128) void qkv_proj_kernel(
    const float* __restrict__ hs, const _Float16* __restrict__ Wt,
    const float* __restrict__ bqkv,
    _Float16* __restrict__ qws, _Float16* __restrict__ kws,
    _Float16* __restrict__ vtws) {
  const int lane = threadIdx.x & 31;
  const int ln   = lane & 15;
  const bool hi  = lane >= 16;
  const int tile = blockIdx.x * 4 + (threadIdx.x >> 5);
  const int nT   = (3 * DMODEL) / 64;          // 48
  const int mT   = tile / nT;                  // 0..127 (32 rows each)
  const int n0   = (tile % nT) * 64;

  v8f acc[2][4] = {};
  const float* rowp0 = hs + (size_t)(mT * 32 + ln) * DMODEL;
  const float* rowp1 = rowp0 + (size_t)16 * DMODEL;
  for (int k0 = 0; k0 < DMODEL; k0 += 32) {
    const v16h a0 = load_a_f32cvt(rowp0, k0, hi);
    const v16h a1 = load_a_f32cvt(rowp1, k0, hi);
#pragma unroll
    for (int t = 0; t < 4; ++t) {
      const v16h b = load_b_kmajor(Wt, DMODEL, n0 + t * 16 + ln, k0, hi);
      acc[0][t] = WMMA_F16(a0, b, acc[0][t]);
      acc[1][t] = WMMA_F16(a1, b, acc[1][t]);
    }
  }
  // Scatter into head-major Q/K and transposed V (all f16).
#pragma unroll
  for (int u = 0; u < 2; ++u) {
#pragma unroll
    for (int t = 0; t < 4; ++t) {
      const int colg = n0 + t * 16 + ln;
      const float bias = bqkv[colg];
#pragma unroll
      for (int m = 0; m < 8; ++m) {
        const int rowg = mT * 32 + u * 16 + m + (hi ? 8 : 0);
        const int bb = rowg >> 11;             // / SEQ
        const int s  = rowg & (SEQ - 1);
        const _Float16 v = (_Float16)(acc[u][t][m] + bias);
        if (colg < DMODEL) {
          const int dm = colg, h = dm >> 6, d = dm & 63;
          qws[((((size_t)bb * NHEADS + h) * SEQ) + s) * HEADDIM + d] = v;
        } else if (colg < 2 * DMODEL) {
          const int dm = colg - DMODEL, h = dm >> 6, d = dm & 63;
          kws[((((size_t)bb * NHEADS + h) * SEQ) + s) * HEADDIM + d] = v;
        } else {
          const int dm = colg - 2 * DMODEL, h = dm >> 6, d = dm & 63;
          vtws[((((size_t)bb * NHEADS + h) * HEADDIM) + d) * SEQ + s] = v;
        }
      }
    }
  }
}

// ---------------------------------------------------------------------------
// Kernel 2: fused scores -> mask/scale -> softmax -> weights out -> PV.
// One wave per (b,h, 16-row block). LDS: 16 x 2048 f32 scores (128 KB).
// ---------------------------------------------------------------------------
__global__ __launch_bounds__(32) void attn_fused_kernel(
    const _Float16* __restrict__ qws, const _Float16* __restrict__ kws,
    const _Float16* __restrict__ vtws,
    float* __restrict__ wout,            // attn_weights [B,H,S,S]
    _Float16* __restrict__ aows) {       // attn_out f16 [B,S,D]
  extern __shared__ float sc[];          // [16][SEQ]
  const int lane = threadIdx.x;
  const int ln   = lane & 15;
  const bool hi  = lane >= 16;
  const int bid  = blockIdx.x;           // bh*128 + rb
  const int rb   = bid & 127;
  const int bh   = bid >> 7;
  const int row0 = rb * 16;

  const _Float16* Q  = qws  + (size_t)bh * SEQ * HEADDIM;
  const _Float16* K  = kws  + (size_t)bh * SEQ * HEADDIM;
  const _Float16* Vt = vtws + (size_t)bh * HEADDIM * SEQ;

  // ---- scores = (Q K^T)/8 with causal mask, into LDS ----
  const _Float16* qrow = Q + (size_t)(row0 + ln) * HEADDIM;
  for (int c0 = 0; c0 < SEQ; c0 += 16) {
    if (c0 <= row0 + 15) {
      v8f acc = {};
#pragma unroll
      for (int k0 = 0; k0 < HEADDIM; k0 += 32) {
        const v16h a = load_a_f16(qrow, k0, hi);
        const v16h b = load_b_kmajor(K, HEADDIM, c0 + ln, k0, hi);
        acc = WMMA_F16(a, b, acc);
      }
#pragma unroll
      for (int m = 0; m < 8; ++m) {
        const int r  = m + (hi ? 8 : 0);
        const int rg = row0 + r, cg = c0 + ln;
        sc[r * SEQ + cg] = (cg <= rg) ? acc[m] * 0.125f : -10000.0f;
      }
    } else {  // fully masked tile
#pragma unroll
      for (int m = 0; m < 8; ++m) sc[(m + (hi ? 8 : 0)) * SEQ + c0 + ln] = -10000.0f;
    }
  }
  __syncthreads();

  // ---- per-row softmax (wave-parallel over columns) + write weights ----
  for (int r = 0; r < 16; ++r) {
    float* srow = sc + r * SEQ;
    float mx = -3.0e38f;
    for (int i = lane; i < SEQ; i += 32) mx = fmaxf(mx, srow[i]);
#pragma unroll
    for (int off = 16; off >= 1; off >>= 1) mx = fmaxf(mx, __shfl_xor(mx, off, 32));
    float sum = 0.0f;
    for (int i = lane; i < SEQ; i += 32) {
      const float e = __expf(srow[i] - mx);
      srow[i] = e;
      sum += e;
    }
#pragma unroll
    for (int off = 16; off >= 1; off >>= 1) sum += __shfl_xor(sum, off, 32);
    const float inv = 1.0f / sum;
    float* grow = wout + ((size_t)bh * SEQ + row0 + r) * SEQ;
    for (int i = lane; i < SEQ; i += 32) {
      const float p = srow[i] * inv;
      srow[i] = p;
      grow[i] = p;
    }
  }
  __syncthreads();

  // ---- PV: out(16x64) = P(16x[0..row0+16)) x V ----
  v8f acc[4] = {};
  const int kend = row0 + 16;            // causal: later columns have weight 0
  const float* prow = sc + (size_t)ln * SEQ;
  for (int k0 = 0; k0 < kend; k0 += 32) {
    v16h a;
    const int o = k0 + (hi ? 8 : 0);
#pragma unroll
    for (int j = 0; j < 8; ++j) {
      a[j]     = (_Float16)prow[o + j];
      a[j + 8] = (_Float16)prow[o + 16 + j];
    }
#pragma unroll
    for (int t = 0; t < 4; ++t) {
      const v16h b = load_b_kmajor(Vt, SEQ, t * 16 + ln, k0, hi);
      acc[t] = WMMA_F16(a, b, acc[t]);
    }
  }
  const int b = bh >> 4, h = bh & 15;
#pragma unroll
  for (int t = 0; t < 4; ++t) {
#pragma unroll
    for (int m = 0; m < 8; ++m) {
      const int s = row0 + m + (hi ? 8 : 0);
      aows[(((size_t)b * SEQ + s) * DMODEL) + h * HEADDIM + t * 16 + ln] =
          (_Float16)acc[t][m];
    }
  }
}

// ---------------------------------------------------------------------------
// Kernel 3: output projection. M=4096, N=1024, K=1024. One wave -> 32x64 tile.
// ---------------------------------------------------------------------------
__global__ __launch_bounds__(128) void out_proj_kernel(
    const _Float16* __restrict__ ao, const _Float16* __restrict__ Wt,
    const float* __restrict__ bp, float* __restrict__ out) {
  const int lane = threadIdx.x & 31;
  const int ln   = lane & 15;
  const bool hi  = lane >= 16;
  const int tile = blockIdx.x * 4 + (threadIdx.x >> 5);
  const int nT   = DMODEL / 64;          // 16
  const int mT   = tile / nT;            // 0..127 (32 rows each)
  const int n0   = (tile % nT) * 64;

  v8f acc[2][4] = {};
  const _Float16* rowp0 = ao + (size_t)(mT * 32 + ln) * DMODEL;
  const _Float16* rowp1 = rowp0 + (size_t)16 * DMODEL;
  for (int k0 = 0; k0 < DMODEL; k0 += 32) {
    const v16h a0 = load_a_f16(rowp0, k0, hi);
    const v16h a1 = load_a_f16(rowp1, k0, hi);
#pragma unroll
    for (int t = 0; t < 4; ++t) {
      const v16h b = load_b_kmajor(Wt, DMODEL, n0 + t * 16 + ln, k0, hi);
      acc[0][t] = WMMA_F16(a0, b, acc[0][t]);
      acc[1][t] = WMMA_F16(a1, b, acc[1][t]);
    }
  }
#pragma unroll
  for (int u = 0; u < 2; ++u) {
#pragma unroll
    for (int t = 0; t < 4; ++t) {
      const int colg = n0 + t * 16 + ln;
      const float bias = bp[colg];
#pragma unroll
      for (int m = 0; m < 8; ++m) {
        const int rowg = mT * 32 + u * 16 + m + (hi ? 8 : 0);
        out[(size_t)rowg * DMODEL + colg] = acc[u][t][m] + bias;
      }
    }
  }
}

// ---------------------------------------------------------------------------
extern "C" void kernel_launch(void* const* d_in, const int* in_sizes, int n_in,
                              void* d_out, int out_size, void* d_ws, size_t ws_size,
                              hipStream_t stream) {
  const float* hs   = (const float*)d_in[0];
  const float* Wqkv = (const float*)d_in[1];
  const float* bqkv = (const float*)d_in[2];
  const float* Wp   = (const float*)d_in[3];
  const float* bp   = (const float*)d_in[4];

  float* out_attn    = (float*)d_out;                              // [B,S,D]
  float* out_weights = (float*)d_out + (size_t)MROWS * DMODEL;     // [B,H,S,S]

  char* ws = (char*)d_ws;
  const size_t seg = (size_t)MROWS * DMODEL * sizeof(_Float16);    // 8 MB
  _Float16* qws    = (_Float16*)(ws + 0 * seg);
  _Float16* kws    = (_Float16*)(ws + 1 * seg);
  _Float16* vtws   = (_Float16*)(ws + 2 * seg);
  _Float16* aows   = (_Float16*)(ws + 3 * seg);
  _Float16* wqkvt  = (_Float16*)(ws + 4 * seg);                    // 6 MB
  _Float16* wpt    = (_Float16*)(ws + 4 * seg + (size_t)3 * DMODEL * DMODEL * 2);

  // K0: weight transpose+convert (f32 [K][N] -> f16 [N][K])
  dim3 tb(64, 4);
  wprep_kernel<<<dim3((3 * DMODEL) / 64, DMODEL / 64), tb, 0, stream>>>(
      Wqkv, wqkvt, 3 * DMODEL, DMODEL);
  wprep_kernel<<<dim3(DMODEL / 64, DMODEL / 64), tb, 0, stream>>>(
      Wp, wpt, DMODEL, DMODEL);

  // K1: (4096/32)*(3072/64) = 6144 waves -> 1536 blocks of 4 waves
  qkv_proj_kernel<<<1536, 128, 0, stream>>>(hs, wqkvt, bqkv, qws, kws, vtws);

  // K2: B*H*(S/16) = 4096 single-wave blocks, 128 KB dynamic LDS each
  const int lds_bytes = 16 * SEQ * (int)sizeof(float);             // 131072
  hipFuncSetAttribute((const void*)attn_fused_kernel,
                      hipFuncAttributeMaxDynamicSharedMemorySize, lds_bytes);
  attn_fused_kernel<<<BH * (SEQ / 16), 32, lds_bytes, stream>>>(
      qws, kws, vtws, out_weights, aows);

  // K3: (4096/32)*(1024/64) = 2048 waves -> 512 blocks of 4 waves
  out_proj_kernel<<<512, 128, 0, stream>>>(aows, wpt, bp, out_attn);
}
